// TransformerBlock_28999619183040
// MI455X (gfx1250) — matrix operand order repaired and checked by reading, not compiled, not run on hardware
//
#include <hip/hip_runtime.h>

// ---------------- types ----------------
typedef __bf16 bf16;
typedef __attribute__((ext_vector_type(16))) __bf16 v16bf;
typedef __attribute__((ext_vector_type(8)))  __bf16 v8bf;
typedef __attribute__((ext_vector_type(8)))  float  v8f;
typedef int v4i __attribute__((vector_size(16)));

#define AS1 __attribute__((address_space(1)))
#define AS3 __attribute__((address_space(3)))

// async global->LDS path (gfx1250), guarded so the file compiles either way
#if defined(__has_builtin)
#if __has_builtin(__builtin_amdgcn_global_load_async_to_lds_b128) && \
    __has_builtin(__builtin_amdgcn_s_wait_asynccnt)
#define HAVE_ASYNC_LDS 1
#endif
#endif
#ifndef HAVE_ASYNC_LDS
#define HAVE_ASYNC_LDS 0
#endif

union Frag16 { v16bf v; v8bf h[2]; };

__device__ __forceinline__ v8f wmma_bf16(v16bf a, v16bf b, v8f c) {
    return __builtin_amdgcn_wmma_f32_16x16x32_bf16(
        /*neg_a=*/false, a, /*neg_b=*/false, b,
        /*c_mod=*/(short)0, c, /*reuse_a=*/false, /*reuse_b=*/false);
}

// copy 16 bf16 (32B) global -> LDS, async when available
__device__ __forceinline__ void cp16_g2l(const bf16* g, bf16* l) {
#if HAVE_ASYNC_LDS
    __builtin_amdgcn_global_load_async_to_lds_b128(
        (AS1 v4i*)(g),     (AS3 v4i*)(l),     0, 0);
    __builtin_amdgcn_global_load_async_to_lds_b128(
        (AS1 v4i*)(g + 8), (AS3 v4i*)(l + 8), 0, 0);
#else
    *(uint4*)l       = *(const uint4*)g;
    *(uint4*)(l + 8) = *(const uint4*)(g + 8);
#endif
}

// A fragment (16x32 bf16, MxK). lane<16 -> M=lane, K {0..7,16..23};
// lane>=16 -> M=lane-16, K {8..15,24..31}.
__device__ __forceinline__ v16bf load_a_frag(const bf16* p, int ld) {
    int lane = threadIdx.x & 31;
    int r    = lane & 15;
    int koff = (lane < 16) ? 0 : 8;
    const bf16* q = p + r * ld + koff;
    Frag16 f;
    f.h[0] = *(const v8bf*)(q);
    f.h[1] = *(const v8bf*)(q + 16);
    return f.v;
}

// B fragment (32x16 bf16, KxN), source row-major [N][K].
// lane<16 -> N=lane, K {0..7,16..23}; lane>=16 -> N=lane-16, K {8..15,24..31}.
__device__ __forceinline__ v16bf load_b_frag(const bf16* p, int ld) {
    int lane = threadIdx.x & 31;
    int n    = lane & 15;
    int koff = (lane < 16) ? 0 : 8;
    const bf16* q = p + n * ld + koff;
    Frag16 f;
    f.h[0] = *(const v8bf*)(q);
    f.h[1] = *(const v8bf*)(q + 16);
    return f.v;
}

// ---------------- constants ----------------
#define NB   2
#define NN   2048
#define ND   1024
#define NH   16
#define NDH  64
#define NDFF 4096
#define NTOK (NB * NN)   // 4096

// ---------------- cast ----------------
__global__ void cast_bf16_kernel(const float* __restrict__ in,
                                 bf16* __restrict__ out, int n) {
    int i = blockIdx.x * blockDim.x + threadIdx.x;
    if (i < n) out[i] = (bf16)in[i];
}

// ---------------- layernorm (row = 1024) -> bf16 ----------------
__global__ __launch_bounds__(256)
void ln_bf16_kernel(const float* __restrict__ x, const float* __restrict__ gamma,
                    bf16* __restrict__ out) {
    __shared__ float red[16];
    int row = blockIdx.x;
    const float* xr = x + (size_t)row * ND;
    float v[4], sum = 0.f, sq = 0.f;
    for (int i = 0; i < 4; ++i) {
        v[i] = xr[threadIdx.x + i * 256];
        sum += v[i];
        sq  += v[i] * v[i];
    }
    for (int off = 16; off > 0; off >>= 1) {
        sum += __shfl_xor(sum, off, 32);
        sq  += __shfl_xor(sq,  off, 32);
    }
    int wave = threadIdx.x >> 5, lane = threadIdx.x & 31;
    if (lane == 0) { red[wave] = sum; red[8 + wave] = sq; }
    __syncthreads();
    float ts = 0.f, tq = 0.f;
    for (int w = 0; w < 8; ++w) { ts += red[w]; tq += red[8 + w]; }
    float mu  = ts * (1.f / ND);
    float var = tq * (1.f / ND) - mu * mu;
    float rs  = rsqrtf(var + 1e-5f);
    for (int i = 0; i < 4; ++i) {
        int c = threadIdx.x + i * 256;
        out[(size_t)row * ND + c] = (bf16)((v[i] - mu) * rs * gamma[c]);
    }
}

// ---------------- GEMM: C[M,Nc] = A[M,K](bf16) * W[Nc,K](bf16)^T ----------------
// 128x128 tile per 256-thread WG, 8 waves each 64x32, K step 32,
// double-buffered LDS fed by async global->LDS copies.
template <bool BIAS, bool RES, bool OUTBF>
__global__ __launch_bounds__(256)
void gemm_bf16(const bf16* __restrict__ A, const bf16* __restrict__ W,
               const float* __restrict__ bias, const float* __restrict__ res,
               float* __restrict__ outF, bf16* __restrict__ outB,
               int M, int Nc, int K) {
    __shared__ bf16 As[2][128][32];
    __shared__ bf16 Bs[2][128][32];
    int m0 = blockIdx.y * 128;
    int n0 = blockIdx.x * 128;
    int tid = threadIdx.x;
    int wave = tid >> 5;
    int wm = (wave >> 2) * 64;   // 0 / 64
    int wn = (wave & 3) * 32;    // 0 / 32 / 64 / 96
    // cooperative tile copy assignment: thread -> (row, 16-elem column chunk)
    int lrow = tid >> 1;
    int lcol = (tid & 1) * 16;
    const bf16* gA = &A[(size_t)(m0 + lrow) * K + lcol];
    const bf16* gB = &W[(size_t)(n0 + lrow) * K + lcol];

    v8f acc[4][2] = {};
    int nk = K / 32;
    // prologue: stage 0
    cp16_g2l(gA, &As[0][lrow][lcol]);
    cp16_g2l(gB, &Bs[0][lrow][lcol]);
    for (int i = 0; i < nk; ++i) {
        int buf = i & 1;
        if (i + 1 < nk) {
            cp16_g2l(gA + (i + 1) * 32, &As[buf ^ 1][lrow][lcol]);
            cp16_g2l(gB + (i + 1) * 32, &Bs[buf ^ 1][lrow][lcol]);
        }
#if HAVE_ASYNC_LDS
        if (i + 1 < nk) __builtin_amdgcn_s_wait_asynccnt(4);
        else            __builtin_amdgcn_s_wait_asynccnt(0);
#endif
        __syncthreads();
        v16bf bfr0 = load_b_frag(&Bs[buf][wn][0], 32);
        v16bf bfr1 = load_b_frag(&Bs[buf][wn + 16][0], 32);
        for (int mi = 0; mi < 4; ++mi) {
            v16bf afr = load_a_frag(&As[buf][wm + mi * 16][0], 32);
            acc[mi][0] = wmma_bf16(afr, bfr0, acc[mi][0]);
            acc[mi][1] = wmma_bf16(afr, bfr1, acc[mi][1]);
        }
        __syncthreads();
    }
    int lane = tid & 31;
    int rhalf = (lane < 16) ? 0 : 8;
    for (int mi = 0; mi < 4; ++mi) {
        int rowb = m0 + wm + mi * 16 + rhalf;
        for (int ni = 0; ni < 2; ++ni) {
            int n = n0 + wn + ni * 16 + (lane & 15);
            float bia = BIAS ? bias[n] : 0.0f;
            for (int r = 0; r < 8; ++r) {
                size_t idx = (size_t)(rowb + r) * Nc + n;
                float v = acc[mi][ni][r] + bia;
                if (RES) v += res[idx];
                if (OUTBF) outB[idx] = (bf16)v;
                else       outF[idx] = v;
            }
        }
    }
}

// ---------------- RoPE + head split: qkv f32 -> Q,K,V bf16 [B,H,N,DH] ----------------
__global__ void rope_kernel(const float* __restrict__ qkv, const float* __restrict__ freqs,
                            bf16* __restrict__ Qo, bf16* __restrict__ Ko,
                            bf16* __restrict__ Vo) {
    int idx = blockIdx.x * blockDim.x + threadIdx.x;
    if (idx >= NB * NN * NH * 32) return;
    int d = idx & 31;
    int h = (idx >> 5) & 15;
    int n = (idx >> 9) & 2047;
    int b = idx >> 20;
    size_t tok = (size_t)b * NN + n;
    const float* base = qkv + tok * (3 * ND) + h * NDH;
    float fr = freqs[n * NDH + d];          // freqs[n][d] == freqs[n][d+32]
    float c = __cosf(fr), s = __sinf(fr);
    float q1 = base[d],         q2 = base[d + 32];
    float k1 = base[ND + d],    k2 = base[ND + d + 32];
    size_t o = ((size_t)(b * NH + h) * NN + n) * NDH;
    Qo[o + d]      = (bf16)(q1 * c - q2 * s);
    Qo[o + d + 32] = (bf16)(q2 * c + q1 * s);
    Ko[o + d]      = (bf16)(k1 * c - k2 * s);
    Ko[o + d + 32] = (bf16)(k2 * c + k1 * s);
    Vo[o + d]      = (bf16)base[2 * ND + d];
    Vo[o + d + 32] = (bf16)base[2 * ND + d + 32];
}

// ---------------- flash attention ----------------
// grid: (N/128, B*H). WG = 256 threads = 8 waves; wave handles 16 query rows.
__global__ __launch_bounds__(256)
void attn_kernel(const bf16* __restrict__ Q, const bf16* __restrict__ Kk,
                 const bf16* __restrict__ V, const unsigned char* __restrict__ mask,
                 bf16* __restrict__ Obf) {
    __shared__ bf16 Ks[64][64];          // [key][dh]
    __shared__ bf16 Vt[64][64];          // [dh][key]
    __shared__ bf16 Ps[8][16][64];       // per-wave P staging [row][key]
    int bh = blockIdx.y;
    int b  = bh >> 4;
    int h  = bh & 15;
    int mblk = blockIdx.x;
    int tid = threadIdx.x, wave = tid >> 5, lane = tid & 31;
    int rowbase = mblk * 128 + wave * 16;
    int rhalf = (lane < 16) ? 0 : 8;

    // Q fragments straight from global (row-major [n][dh])
    const bf16* qbase = Q + ((size_t)bh * NN + rowbase) * NDH;
    v16bf qa[2];
    {
        int r = lane & 15;
        int koff = (lane < 16) ? 0 : 8;
        for (int c = 0; c < 2; ++c) {
            Frag16 f;
            const bf16* p = qbase + r * NDH + 32 * c + koff;
            f.h[0] = *(const v8bf*)p;
            f.h[1] = *(const v8bf*)(p + 16);
            qa[c] = f.v;
        }
    }

    v8f o[4] = {};
    float mrow[8], lrow[8];
    for (int r = 0; r < 8; ++r) { mrow[r] = -1e30f; lrow[r] = 0.f; }
    const float scale = 0.125f;          // 1/sqrt(64)

    int nkb = mblk * 2 + 2;              // causal: key blocks with j0 <= max row
    for (int jb = 0; jb < nkb; ++jb) {
        int j0 = jb * 64;
        __syncthreads();
        {   // cooperative load: K tile async, V tile transposed manually
            int row = tid >> 2, col = (tid & 3) * 16;
            const bf16* ksrc = Kk + ((size_t)bh * NN + j0 + row) * NDH + col;
            cp16_g2l(ksrc, &Ks[row][col]);
            const bf16* vsrc = V + ((size_t)bh * NN + j0 + row) * NDH + col;
            for (int e = 0; e < 16; ++e) Vt[col + e][row] = vsrc[e];
        }
#if HAVE_ASYNC_LDS
        __builtin_amdgcn_s_wait_asynccnt(0);
#endif
        __syncthreads();
        if (j0 <= rowbase + 15) {
            bool fullblk = (j0 + 63 <= rowbase);   // whole block causally valid
            // S = Q * K^T  (4 key sub-tiles of 16)
            float s[4][8];
            for (int nt = 0; nt < 4; ++nt) {
                v8f sa = {};
                for (int c = 0; c < 2; ++c) {
                    v16bf bfr = load_b_frag(&Ks[nt * 16][32 * c], 64);
                    sa = wmma_bf16(qa[c], bfr, sa);
                }
                int keyg = j0 + nt * 16 + (lane & 15);
                bool mk = mask[b * NN + keyg] != 0;
                if (fullblk) {
                    for (int r = 0; r < 8; ++r)
                        s[nt][r] = mk ? sa[r] * scale : -1e30f;
                } else {
                    for (int r = 0; r < 8; ++r) {
                        int rowg = rowbase + rhalf + r;
                        s[nt][r] = (mk && keyg <= rowg) ? sa[r] * scale : -1e30f;
                    }
                }
            }
            // online softmax per row (rows live in one half-wave: xor 1,2,4,8)
            float corr[8];
            for (int r = 0; r < 8; ++r) {
                float mx = fmaxf(fmaxf(s[0][r], s[1][r]), fmaxf(s[2][r], s[3][r]));
                for (int off = 1; off < 16; off <<= 1)
                    mx = fmaxf(mx, __shfl_xor(mx, off, 32));
                float mnew = fmaxf(mrow[r], mx);
                float c0 = __expf(mrow[r] - mnew);
                float sm = 0.f;
                for (int nt = 0; nt < 4; ++nt) {
                    float p = __expf(s[nt][r] - mnew);
                    s[nt][r] = p;
                    sm += p;
                }
                for (int off = 1; off < 16; off <<= 1)
                    sm += __shfl_xor(sm, off, 32);
                lrow[r] = lrow[r] * c0 + sm;
                mrow[r] = mnew;
                corr[r] = c0;
            }
            for (int d = 0; d < 4; ++d)
                for (int r = 0; r < 8; ++r)
                    o[d][r] *= corr[r];
            // spill P (C-layout) to LDS, reload as A fragments
            for (int nt = 0; nt < 4; ++nt)
                for (int r = 0; r < 8; ++r)
                    Ps[wave][rhalf + r][nt * 16 + (lane & 15)] = (bf16)s[nt][r];
            asm volatile("s_wait_dscnt 0" ::: "memory");
            // O += P * V
            for (int c = 0; c < 2; ++c) {
                v16bf pa = load_a_frag(&Ps[wave][0][0] + 32 * c, 64);
                for (int d = 0; d < 4; ++d) {
                    v16bf vb = load_b_frag(&Vt[d * 16][32 * c], 64);
                    o[d] = wmma_bf16(pa, vb, o[d]);
                }
            }
        }
    }
    // normalize and write bf16 to [tok][D] with head-major columns
    for (int d = 0; d < 4; ++d)
        for (int r = 0; r < 8; ++r) {
            int n = rowbase + rhalf + r;
            int col = d * 16 + (lane & 15);
            float v = o[d][r] / lrow[r];
            Obf[((size_t)b * NN + n) * ND + h * NDH + col] = (bf16)v;
        }
}

// ---------------- SiLU-GLU elementwise ----------------
__global__ void glu_kernel(const bf16* __restrict__ up, const bf16* __restrict__ gate,
                           bf16* __restrict__ out, int n) {
    int i = blockIdx.x * blockDim.x + threadIdx.x;
    if (i < n) {
        float u = (float)up[i], g = (float)gate[i];
        out[i] = (bf16)(u * (g / (1.f + __expf(-g))));
    }
}

// ---------------- launch ----------------
extern "C" void kernel_launch(void* const* d_in, const int* in_sizes, int n_in,
                              void* d_out, int out_size, void* d_ws, size_t ws_size,
                              hipStream_t stream) {
    const float* x         = (const float*)d_in[0];
    const float* gamma_pre = (const float*)d_in[1];
    const float* w_qkv     = (const float*)d_in[2];
    const float* w_o       = (const float*)d_in[3];
    const float* gamma_ff  = (const float*)d_in[4];
    const float* w_up      = (const float*)d_in[5];
    const float* b_up      = (const float*)d_in[6];
    const float* w_gate    = (const float*)d_in[7];
    const float* b_gate    = (const float*)d_in[8];
    const float* w_out     = (const float*)d_in[9];
    const float* b_out     = (const float*)d_in[10];
    const float* freqs     = (const float*)d_in[11];
    const unsigned char* mask = (const unsigned char*)d_in[12];
    float* out = (float*)d_out;

    char* ws = (char*)d_ws;
    size_t off = 0;
    auto alloc = [&](size_t bytes) -> void* {
        void* p = ws + off;
        off += (bytes + 255) & ~(size_t)255;
        return p;
    };
    bf16* wqkv_b  = (bf16*)alloc((size_t)3 * ND * ND * 2);
    bf16* wo_b    = (bf16*)alloc((size_t)ND * ND * 2);
    bf16* wup_b   = (bf16*)alloc((size_t)NDFF * ND * 2);
    bf16* wgate_b = (bf16*)alloc((size_t)NDFF * ND * 2);
    bf16* wout_b  = (bf16*)alloc((size_t)ND * NDFF * 2);
    bf16* h_b     = (bf16*)alloc((size_t)NTOK * ND * 2);
    float* qkv_f  = (float*)alloc((size_t)NTOK * 3 * ND * 4);
    bf16* q_b     = (bf16*)alloc((size_t)NTOK * ND * 2);
    bf16* k_b     = (bf16*)alloc((size_t)NTOK * ND * 2);
    bf16* v_b     = (bf16*)alloc((size_t)NTOK * ND * 2);
    bf16* o_b     = (bf16*)alloc((size_t)NTOK * ND * 2);
    float* x1_f   = (float*)alloc((size_t)NTOK * ND * 4);
    bf16* h2_b    = (bf16*)alloc((size_t)NTOK * ND * 2);
    bf16* up_b    = (bf16*)alloc((size_t)NTOK * NDFF * 2);
    bf16* gate_b  = (bf16*)alloc((size_t)NTOK * NDFF * 2);
    bf16* ffh_b   = (bf16*)alloc((size_t)NTOK * NDFF * 2);

    // 1) weight casts
    {
        int n;
        n = 3 * ND * ND;  cast_bf16_kernel<<<(n + 255) / 256, 256, 0, stream>>>(w_qkv, wqkv_b, n);
        n = ND * ND;      cast_bf16_kernel<<<(n + 255) / 256, 256, 0, stream>>>(w_o, wo_b, n);
        n = NDFF * ND;    cast_bf16_kernel<<<(n + 255) / 256, 256, 0, stream>>>(w_up, wup_b, n);
        n = NDFF * ND;    cast_bf16_kernel<<<(n + 255) / 256, 256, 0, stream>>>(w_gate, wgate_b, n);
        n = ND * NDFF;    cast_bf16_kernel<<<(n + 255) / 256, 256, 0, stream>>>(w_out, wout_b, n);
    }
    // 2) pre-norm
    ln_bf16_kernel<<<NTOK, 256, 0, stream>>>(x, gamma_pre, h_b);
    // 3) qkv = h @ w_qkv^T  (f32 out)
    gemm_bf16<false, false, false><<<dim3(3 * ND / 128, NTOK / 128), 256, 0, stream>>>(
        h_b, wqkv_b, nullptr, nullptr, qkv_f, nullptr, NTOK, 3 * ND, ND);
    // 4) rope + head split -> bf16 Q,K,V [B,H,N,DH]
    {
        int n = NB * NN * NH * 32;
        rope_kernel<<<(n + 255) / 256, 256, 0, stream>>>(qkv_f, freqs, q_b, k_b, v_b);
    }
    // 5) flash attention -> o_b [tok][D]
    attn_kernel<<<dim3(NN / 128, NB * NH), 256, 0, stream>>>(q_b, k_b, v_b, mask, o_b);
    // 6) x1 = x + o @ w_o^T
    gemm_bf16<false, true, false><<<dim3(ND / 128, NTOK / 128), 256, 0, stream>>>(
        o_b, wo_b, nullptr, x, x1_f, nullptr, NTOK, ND, ND);
    // 7) ff pre-norm
    ln_bf16_kernel<<<NTOK, 256, 0, stream>>>(x1_f, gamma_ff, h2_b);
    // 8) up / gate (bf16 out, fused bias)
    gemm_bf16<true, false, true><<<dim3(NDFF / 128, NTOK / 128), 256, 0, stream>>>(
        h2_b, wup_b, b_up, nullptr, nullptr, up_b, NTOK, NDFF, ND);
    gemm_bf16<true, false, true><<<dim3(NDFF / 128, NTOK / 128), 256, 0, stream>>>(
        h2_b, wgate_b, b_gate, nullptr, nullptr, gate_b, NTOK, NDFF, ND);
    // 9) ffh = up * silu(gate)
    {
        int n = NTOK * NDFF;
        glu_kernel<<<(n + 255) / 256, 256, 0, stream>>>(up_b, gate_b, ffh_b, n);
    }
    // 10) out = x1 + ffh @ w_out^T + b_out  (f32 out)
    gemm_bf16<true, true, false><<<dim3(ND / 128, NTOK / 128), 256, 0, stream>>>(
        ffh_b, wout_b, b_out, x1_f, out, nullptr, NTOK, ND, NDFF);
}
